// DMoELinear_35622458753638
// MI455X (gfx1250) — matrix operand order repaired and checked by reading, not compiled, not run on hardware
//
#include <hip/hip_runtime.h>
#include <hip/hip_bf16.h>

typedef __bf16 bf16;
typedef bf16 v16bf __attribute__((ext_vector_type(16)));
typedef bf16 v8bf  __attribute__((ext_vector_type(8)));
typedef bf16 v4bf  __attribute__((ext_vector_type(4)));
typedef float v8f  __attribute__((ext_vector_type(8)));
typedef float v4f  __attribute__((ext_vector_type(4)));

#define M_TILE   128
#define N_TILE   128
#define K_TILE   64
#define K_DIM    1024
#define N_OUT    2048
#define KT_ITERS (K_DIM / K_TILE)
#define LDS_STR  72   // 64 bf16 + 8 pad = 144 bytes/row, 16B aligned

__global__ __launch_bounds__(256)
void moe_gemm_bf16_wmma(const float* __restrict__ x,
                        const float* __restrict__ w,
                        const float* __restrict__ bias,
                        const int*   __restrict__ ids,
                        bf16*        __restrict__ out)
{
    __shared__ bf16 As[M_TILE * LDS_STR];
    __shared__ bf16 Bs[N_TILE * LDS_STR];
    __shared__ int  sIds[M_TILE];

    const int tid    = threadIdx.x;
    const int lane   = tid & 31;
    const int wv     = tid >> 5;       // wave 0..7 -> 16-row stripe
    const int laneLo = lane & 15;
    const int hi     = lane >> 4;      // lane half

    const int rowBase = blockIdx.x * M_TILE;
    const int colBase = blockIdx.y * N_TILE;

    if (tid < M_TILE) sIds[tid] = ids[rowBase + tid];
    __syncthreads();

    const int eLo = sIds[0];
    const int eHi = sIds[M_TILE - 1];

    v8f acc[8];
#pragma unroll
    for (int n = 0; n < 8; ++n) {
        v8f z = {0.f, 0.f, 0.f, 0.f, 0.f, 0.f, 0.f, 0.f};
        acc[n] = z;
    }

    for (int e = eLo; e <= eHi; ++e) {
        const float* wbase = w + ((long)e * N_OUT + colBase) * K_DIM;

        for (int kt = 0; kt < KT_ITERS; ++kt) {
            const int k0 = kt * K_TILE;
            __syncthreads();   // previous compute done before LDS overwrite

            // ---- stage A (expert-masked) and B tiles: fp32 -> bf16 -> LDS ----
#pragma unroll
            for (int j = 0; j < 8; ++j) {
                const int i  = tid + j * 256;      // 0..2047
                const int r  = i >> 4;             // row 0..127
                const int c4 = (i & 15) << 2;      // k offset 0..60 step 4

                const float* ap = x + (long)(rowBase + r) * K_DIM + k0 + c4;
                v4f av = *(const v4f*)ap;
                const float m = (sIds[r] == e) ? 1.0f : 0.0f;
                v4bf ab;
                ab[0] = (bf16)(av[0] * m); ab[1] = (bf16)(av[1] * m);
                ab[2] = (bf16)(av[2] * m); ab[3] = (bf16)(av[3] * m);
                *(v4bf*)&As[r * LDS_STR + c4] = ab;

                const float* bp = wbase + (long)r * K_DIM + k0 + c4;
                v4f bv = *(const v4f*)bp;
                v4bf bb;
                bb[0] = (bf16)bv[0]; bb[1] = (bf16)bv[1];
                bb[2] = (bf16)bv[2]; bb[3] = (bf16)bv[3];
                *(v4bf*)&Bs[r * LDS_STR + c4] = bb;

                // prefetch next K tile toward caches (speculative; dropped if OOB)
                if (kt + 1 < KT_ITERS) {
                    __builtin_prefetch(ap + K_TILE, 0, 3);
                    __builtin_prefetch(bp + K_TILE, 0, 3);
                }
            }
            __syncthreads();

            // ---- 2 WMMA k-steps of 32 over the 64-wide LDS tile ----
#pragma unroll
            for (int ks = 0; ks < 2; ++ks) {
                const int kk = ks * 32;

                // A fragment: lane half h holds K {kk+8h..kk+8h+7, kk+16+8h..}
                const bf16* arow = &As[(wv * 16 + laneLo) * LDS_STR];
                v8bf alo = *(const v8bf*)&arow[kk + hi * 8];
                v8bf ahi = *(const v8bf*)&arow[kk + 16 + hi * 8];
                v16bf afrag = __builtin_shufflevector(
                    alo, ahi, 0,1,2,3,4,5,6,7,8,9,10,11,12,13,14,15);

#pragma unroll
                for (int n = 0; n < 8; ++n) {
                    // B fragment: lane half h holds K {kk+8h..kk+8h+7, kk+16+8h..}
                    const bf16* brow = &Bs[(n * 16 + laneLo) * LDS_STR];
                    v8bf blo = *(const v8bf*)&brow[kk + hi * 8];
                    v8bf bhi = *(const v8bf*)&brow[kk + 16 + hi * 8];
                    v16bf bfrag = __builtin_shufflevector(
                        blo, bhi, 0,1,2,3,4,5,6,7,8,9,10,11,12,13,14,15);

                    acc[n] = __builtin_amdgcn_wmma_f32_16x16x32_bf16(
                        /*neg_a=*/false, afrag, /*neg_b=*/false, bfrag,
                        /*c_mod=*/(short)0, acc[n],
                        /*reuse_a=*/false, /*reuse_b=*/false);
                }
            }
        }
    }

    // ---- epilogue: y.bf16 + bias[ids].bf16, store bf16 ----
    // D layout: VGPR r, lanes 0-15 -> M=r, lanes 16-31 -> M=r+8; N=laneLo
#pragma unroll
    for (int n = 0; n < 8; ++n) {
#pragma unroll
        for (int r = 0; r < 8; ++r) {
            const int mLoc = wv * 16 + r + hi * 8;
            const int row  = rowBase + mLoc;
            const int col  = colBase + n * 16 + laneLo;
            const int eid  = sIds[mLoc];
            const float bv = bias[(long)eid * N_OUT + col];
            const float yb = (float)(bf16)acc[n][r];   // round y to bf16 first
            const float bb = (float)(bf16)bv;          // bias gathered as bf16
            out[(long)row * N_OUT + col] = (bf16)(yb + bb);
        }
    }
}

extern "C" void kernel_launch(void* const* d_in, const int* in_sizes, int n_in,
                              void* d_out, int out_size, void* d_ws, size_t ws_size,
                              hipStream_t stream) {
    (void)n_in; (void)d_ws; (void)ws_size;
    const float* x    = (const float*)d_in[0];   // [N, 1024] fp32
    const float* wgt  = (const float*)d_in[1];   // [8, 2048, 1024] fp32
    const float* bias = (const float*)d_in[2];   // [8, 2048] fp32
    const int*   ids  = (const int*)d_in[3];     // [N] int32, sorted

    const int N = in_sizes[3];                   // 8192
    bf16* out = reinterpret_cast<bf16*>(d_out);  // [N, 2048] bf16

    dim3 grid(N / M_TILE, N_OUT / N_TILE);       // (64, 16)
    dim3 block(256);
    moe_gemm_bf16_wmma<<<grid, block, 0, stream>>>(x, wgt, bias, ids, out);
}
